// LocalDenseSynthesizerAttention_20100446945653
// MI455X (gfx1250) — matrix operand order repaired and checked by reading, not compiled, not run on hardware
//
#include <hip/hip_runtime.h>
#include <hip/hip_bf16.h>

typedef __bf16 v16bf __attribute__((ext_vector_type(16)));
typedef __bf16 v8bf  __attribute__((ext_vector_type(8)));
typedef __bf16 v4bf  __attribute__((ext_vector_type(4)));
typedef float  v8f   __attribute__((ext_vector_type(8)));
typedef float  v4f   __attribute__((ext_vector_type(4)));

#define B_    8
#define T_    2048
#define F_    512
#define H_    8
#define C_    45
#define PAD_  22
#define M_TOT (B_ * T_)   // 16384 token rows
#define NPAD2 384         // H*C = 360 padded to a multiple of 64

union Frag16 { v16bf v; struct { v8bf lo, hi; } h; };
union Acc8   { v8f v; float e[8]; };

// ---------------------------------------------------------------------------
// Weight prep: w [K,N] fp32 row-major -> wT [Npad,K] bf16 (rows >= N zeroed).
// Pre-transposing makes each WMMA B-fragment two contiguous 16B loads per lane.
// ---------------------------------------------------------------------------
__global__ void prep_weight_kernel(const float* __restrict__ w,
                                   __bf16* __restrict__ wT,
                                   int K, int N, int Npad) {
    int idx = blockIdx.x * blockDim.x + threadIdx.x;
    if (idx >= Npad * K) return;
    int n = idx / K;
    int k = idx - n * K;
    float val = (n < N) ? w[k * N + n] : 0.0f;
    wT[idx] = (__bf16)val;
}

// ---------------------------------------------------------------------------
// Tiled GEMM: Out[M, N_TOTAL] = A[M,K] * BT[N,K]^T using v_wmma_f32_16x16x32_bf16.
// Block: 512 threads = 16 waves in a 4x4 (mi,ni) grid; M-block = 64 rows.
// A tile (64x32) staged fp32->bf16 via LDS once per block (LDS stride 40
// halfwords -> conflict-free ds_read_b128 fragment loads).
// ---------------------------------------------------------------------------
template <bool A_BF16, bool RELU, bool OUT_BF16, int N_TOTAL>
__global__ __launch_bounds__(512)
void gemm_kernel(const void* __restrict__ A_raw,
                 const __bf16* __restrict__ BT,
                 void* __restrict__ Out, int K) {
    constexpr int NI  = 4;                 // waves along N
    constexpr int WPN = N_TOTAL / NI;      // columns per wave
    constexpr int NT  = WPN / 16;          // 16-wide tiles per wave
    constexpr int LDA = 40;                // LDS row stride (halfwords), padded

    __shared__ __bf16 Alds[64 * LDA];

    const int tid  = threadIdx.x;
    const int wave = tid >> 5;
    const int lane = tid & 31;
    const int mi   = wave >> 2;            // 0..3
    const int ni   = wave & 3;             // 0..3
    const int g    = lane >> 4;            // half-wave: K-group select
    const int li   = lane & 15;            // row (A) / column (B,C)

    const int m_blk = blockIdx.x * 64;

    v8f acc[NT] = {};

    // cooperative A-tile load mapping: 64 rows x 32 K, 4 elements per thread
    const int ld_row = tid >> 3;           // 0..63
    const int ld_k   = (tid & 7) << 2;     // 0,4,...,28

    for (int k0 = 0; k0 < K; k0 += 32) {
        __syncthreads();
        {
            v4bf pk;
            if (A_BF16) {
                const __bf16* Ab = (const __bf16*)A_raw;
                pk = *(const v4bf*)(Ab + (long)(m_blk + ld_row) * K + k0 + ld_k);
            } else {
                const float* Af = (const float*)A_raw;
                v4f f = *(const v4f*)(Af + (long)(m_blk + ld_row) * K + k0 + ld_k);
                pk[0] = (__bf16)f.x; pk[1] = (__bf16)f.y;
                pk[2] = (__bf16)f.z; pk[3] = (__bf16)f.w;
            }
            *(v4bf*)(&Alds[ld_row * LDA + ld_k]) = pk;
        }
        __syncthreads();

        // A fragment (16x32): lane li = row, halves at K offsets g*8 and 16+g*8
        Frag16 a;
        const __bf16* arow = &Alds[(mi * 16 + li) * LDA];
        a.h.lo = *(const v8bf*)(arow + g * 8);
        a.h.hi = *(const v8bf*)(arow + 16 + g * 8);

#pragma unroll
        for (int nt = 0; nt < NT; ++nt) {
            // B fragment (32x16): lane li = column n, halves at K offsets g*8 and 16+g*8
            const int n = ni * WPN + nt * 16 + li;
            const __bf16* brow = BT + (long)n * K + k0;
            Frag16 b;
            b.h.lo = *(const v8bf*)(brow + g * 8);
            b.h.hi = *(const v8bf*)(brow + 16 + g * 8);
            acc[nt] = __builtin_amdgcn_wmma_f32_16x16x32_bf16(
                false, a.v, false, b.v, (short)0, acc[nt], false, false);
        }
    }

    // C/D layout: element i at row m_base+i, column = lane%16 of tile
    const int m_base = m_blk + mi * 16 + g * 8;
#pragma unroll
    for (int nt = 0; nt < NT; ++nt) {
        const int n = ni * WPN + nt * 16 + li;
        Acc8 c; c.v = acc[nt];
#pragma unroll
        for (int i = 0; i < 8; ++i) {
            float val = c.e[i];
            if (RELU) val = val > 0.0f ? val : 0.0f;
            long off = (long)(m_base + i) * N_TOTAL + n;
            if (OUT_BF16) ((__bf16*)Out)[off] = (__bf16)val;
            else          ((float*)Out)[off]  = val;
        }
    }
}

// ---------------------------------------------------------------------------
// Softmax over each head's C=45 logits. Thread per (token, head); data is
// L2-resident by the time this runs.
// ---------------------------------------------------------------------------
__global__ void softmax_kernel(const float* __restrict__ wgt,
                               __bf16* __restrict__ attn) {
    int idx = blockIdx.x * blockDim.x + threadIdx.x;
    if (idx >= M_TOT * H_) return;
    int t = idx / H_;
    int h = idx - t * H_;
    const float* p = wgt + (long)t * NPAD2 + h * C_;
    float mx = -1e30f;
    for (int c = 0; c < C_; ++c) mx = fmaxf(mx, p[c]);
    float s = 0.0f;
    for (int c = 0; c < C_; ++c) s += __expf(p[c] - mx);
    float inv = 1.0f / s;
    __bf16* q = attn + (long)t * NPAD2 + h * C_;
    for (int c = 0; c < C_; ++c) q[c] = (__bf16)(__expf(p[c] - mx) * inv);
}

// ---------------------------------------------------------------------------
// Banded window mix: x[t, f] = sum_c attn[t, h(f), c] * v[t+c-PAD, f].
// Block per token (512 threads = one per feature) -> coalesced v reads.
// Only ~0.75 GFLOP total, so VALU FMA is ample.
// ---------------------------------------------------------------------------
__global__ __launch_bounds__(512)
void window_kernel(const __bf16* __restrict__ attn,
                   const __bf16* __restrict__ v,
                   __bf16* __restrict__ x) {
    const int bt = blockIdx.x;
    const int f  = threadIdx.x;
    const int b  = bt / T_;
    const int t  = bt - b * T_;
    const int h  = f >> 6;                 // DK = 64
    const __bf16* ap = attn + (long)bt * NPAD2 + h * C_;
    const __bf16* vb = v + (long)b * T_ * F_ + f;
    float acc = 0.0f;
#pragma unroll
    for (int c = 0; c < C_; ++c) {
        int tt = t + c - PAD_;
        if (tt >= 0 && tt < T_)
            acc += (float)ap[c] * (float)vb[(long)tt * F_];
    }
    x[(long)bt * F_ + f] = (__bf16)acc;
}

// ---------------------------------------------------------------------------
extern "C" void kernel_launch(void* const* d_in, const int* in_sizes, int n_in,
                              void* d_out, int out_size, void* d_ws, size_t ws_size,
                              hipStream_t stream) {
    (void)in_sizes; (void)n_in; (void)out_size; (void)ws_size;
    const float* query = (const float*)d_in[0];
    /* d_in[1] = key: unused by the reference computation */
    const float* value = (const float*)d_in[2];
    const float* w1    = (const float*)d_in[3];
    const float* w2    = (const float*)d_in[4];
    const float* w3    = (const float*)d_in[5];
    const float* w_out = (const float*)d_in[6];

    char* ws = (char*)d_ws;
    __bf16* w1T   = (__bf16*)(ws);               // 512*512*2       = 0.50 MB
    __bf16* w2T   = (__bf16*)(ws +   524288);    // 384*512*2       = 0.38 MB
    __bf16* w3T   = (__bf16*)(ws +   917504);    // 512*512*2
    __bf16* woT   = (__bf16*)(ws +  1441792);    // 512*512*2
    __bf16* qrelu = (__bf16*)(ws +  1966080);    // 16384*512*2     = 16 MB
    __bf16* vproj = (__bf16*)(ws + 18743296);    // 16384*512*2     = 16 MB
    float*  wgt   = (float*) (ws + 35520512);    // 16384*384*4     = 25 MB
    __bf16* attn  = (__bf16*)(ws + 60686336);    // 16384*384*2     = 12.6 MB
    __bf16* xbuf  = (__bf16*)(ws + 73269248);    // 16384*512*2     = 16 MB
    // total workspace footprint ~90 MB

    // Stage 0: transpose + bf16-convert the four weight matrices.
    {
        int n1 = 512 * 512;
        int n2 = NPAD2 * 512;
        prep_weight_kernel<<<(n1 + 255) / 256, 256, 0, stream>>>(w1,    w1T, 512, 512,   512);
        prep_weight_kernel<<<(n2 + 255) / 256, 256, 0, stream>>>(w2,    w2T, 512, H_*C_, NPAD2);
        prep_weight_kernel<<<(n1 + 255) / 256, 256, 0, stream>>>(w3,    w3T, 512, 512,   512);
        prep_weight_kernel<<<(n1 + 255) / 256, 256, 0, stream>>>(w_out, woT, 512, 512,   512);
    }

    // Stage 1: q_relu = relu(query @ w1)  (fp32 in, bf16 out)
    gemm_kernel<false, true,  true,  512><<<M_TOT / 64, 512, 0, stream>>>(query, w1T, qrelu, F_);
    // Stage 3: v = value @ w3             (fp32 in, bf16 out)
    gemm_kernel<false, false, true,  512><<<M_TOT / 64, 512, 0, stream>>>(value, w3T, vproj, F_);
    // Stage 2: logits = q_relu @ w2       (bf16 in, fp32 out, N padded 360->384)
    gemm_kernel<true,  false, false, NPAD2><<<M_TOT / 64, 512, 0, stream>>>(qrelu, w2T, wgt, F_);
    // Softmax over C=45 per head
    softmax_kernel<<<(M_TOT * H_ + 255) / 256, 256, 0, stream>>>(wgt, attn);
    // Stage 4: banded window mix -> x (bf16)
    window_kernel<<<M_TOT, 512, 0, stream>>>(attn, vproj, xbuf);
    // Stage 5: out = x @ w_out            (bf16 in, fp32 out to d_out)
    gemm_kernel<true,  false, false, 512><<<M_TOT / 64, 512, 0, stream>>>(xbuf, woT, (float*)d_out, F_);
}
